// CSFlow_33328946217044
// MI455X (gfx1250) — matrix operand order repaired and checked by reading, not compiled, not run on hardware
//
#include <hip/hip_runtime.h>

typedef __attribute__((ext_vector_type(16))) __bf16       v16bf;
typedef __attribute__((ext_vector_type(8)))  float        v8f;
typedef __attribute__((ext_vector_type(8)))  unsigned int v8u;
typedef __attribute__((ext_vector_type(4)))  unsigned int v4u;

constexpr int B_   = 2;
constexpr int D_   = 256;
constexpr int H_   = 48;
constexpr int W_   = 160;
constexpr int HW_  = H_ * W_;    // 7680
constexpr int NPIX = B_ * HW_;   // 15360

// ---------- fp32 -> bf16 split (hi + lo), manual round-to-nearest-even ----------
__device__ __forceinline__ unsigned short f2bf(float f) {
    unsigned u = __builtin_bit_cast(unsigned, f);
    u += 0x7FFFu + ((u >> 16) & 1u);
    return (unsigned short)(u >> 16);
}
__device__ __forceinline__ float bf2f(unsigned short h) {
    unsigned u = ((unsigned)h) << 16;
    return __builtin_bit_cast(float, u);
}

// Split fp32 map [b][c][p] into TRANSPOSED bf16 hi/lo maps [b][p][c]
// (channel-contiguous rows -> WMMA fragments load straight from global, no LDS).
__global__ void split_t_kernel(const float* __restrict__ src,
                               unsigned short* __restrict__ hiT,
                               unsigned short* __restrict__ loT,
                               int total) {
    int idx = blockIdx.x * blockDim.x + threadIdx.x;   // (b*HW + p)*D + c
    if (idx >= total) return;
    int c  = idx & (D_ - 1);
    int bp = idx >> 8;                                 // b*HW + p   (D_ == 256)
    int b  = bp / HW_;
    int p  = bp - b * HW_;
    float x = src[((size_t)b * D_ + c) * HW_ + p];
    unsigned short h = f2bf(x);
    hiT[idx] = h;
    loT[idx] = f2bf(x - bf2f(h));
}

// ---------- WMMA fragment loads straight from transposed global maps ----------
// bf16 A 16x32, contiguous k order used here: lanes<16 elems0-15 -> K0..15; lanes>=16 -> K16..31.
__device__ __forceinline__ v16bf load_a_frag(const unsigned short* p) {
    v4u x = *(const v4u*)(p);          // K base .. +7
    v4u y = *(const v4u*)(p + 8);      // K base+8 .. +15
    v8u r;
    r[0] = x[0]; r[1] = x[1]; r[2] = x[2]; r[3] = x[3];
    r[4] = y[0]; r[5] = y[1]; r[6] = y[2]; r[7] = y[3];
    return __builtin_bit_cast(v16bf, r);
}
// bf16 B 32x16, contiguous k order used here: lane holds column n; lanes<16 -> K0..15, lanes>=16 -> K16..31.
__device__ __forceinline__ v16bf load_b_frag(const unsigned short* p) {
    v8u r = *(const v8u*)(p);          // 16 contiguous K values (32B, aligned)
    return __builtin_bit_cast(v16bf, r);
}

__device__ __forceinline__ v8f wmma_bf16(v16bf a, v16bf b, v8f c) {
    return __builtin_amdgcn_wmma_f32_16x16x32_bf16(false, a, false, b, (short)0, c, false, false);
}

// corr[b,p,q] = (1/16) * sum_c f1[b,c,p]*f2[b,c,q]
// Split-bf16: x = hi + lo; A*B ~= Ah*Bh + Ah*Bl + Al*Bh (fp32 accumulate).
// Wave tile 32x32 (2x2 WMMA tiles); block = 8 waves as 2(M) x 4(N) -> 64x128.
__global__ __launch_bounds__(256)
void corr_wmma_kernel(const unsigned short* __restrict__ f1hiT,
                      const unsigned short* __restrict__ f1loT,
                      const unsigned short* __restrict__ f2hiT,
                      const unsigned short* __restrict__ f2loT,
                      float* __restrict__ out) {
    const int lane = threadIdx.x & 31;
    const int wave = threadIdx.x >> 5;
    const int b    = blockIdx.z;
    const int m0   = blockIdx.y * 64  + (wave >> 2) * 32;
    const int n0   = blockIdx.x * 128 + (wave & 3)  * 32;

    const int hv  = lane >> 4;     // half of wave
    const int l15 = lane & 15;
    const int kbA = hv * 16;
    const int kbB = hv * 16;

    const size_t mapOff = (size_t)b * HW_ * D_;
    const unsigned short* a0h = f1hiT + mapOff + (size_t)(m0 + l15)      * D_ + kbA;
    const unsigned short* a1h = f1hiT + mapOff + (size_t)(m0 + 16 + l15) * D_ + kbA;
    const unsigned short* a0l = f1loT + mapOff + (size_t)(m0 + l15)      * D_ + kbA;
    const unsigned short* a1l = f1loT + mapOff + (size_t)(m0 + 16 + l15) * D_ + kbA;
    const unsigned short* b0h = f2hiT + mapOff + (size_t)(n0 + l15)      * D_ + kbB;
    const unsigned short* b1h = f2hiT + mapOff + (size_t)(n0 + 16 + l15) * D_ + kbB;
    const unsigned short* b0l = f2loT + mapOff + (size_t)(n0 + l15)      * D_ + kbB;
    const unsigned short* b1l = f2loT + mapOff + (size_t)(n0 + 16 + l15) * D_ + kbB;

    v8f acc00 = {}, acc01 = {}, acc10 = {}, acc11 = {};

    #pragma unroll 2
    for (int kk = 0; kk < D_; kk += 32) {
        v16bf A0h = load_a_frag(a0h + kk);
        v16bf A0l = load_a_frag(a0l + kk);
        v16bf A1h = load_a_frag(a1h + kk);
        v16bf A1l = load_a_frag(a1l + kk);
        v16bf B0h = load_b_frag(b0h + kk);
        v16bf B0l = load_b_frag(b0l + kk);
        v16bf B1h = load_b_frag(b1h + kk);
        v16bf B1l = load_b_frag(b1l + kk);

        acc00 = wmma_bf16(A0h, B0h, acc00);
        acc00 = wmma_bf16(A0h, B0l, acc00);
        acc00 = wmma_bf16(A0l, B0h, acc00);

        acc01 = wmma_bf16(A0h, B1h, acc01);
        acc01 = wmma_bf16(A0h, B1l, acc01);
        acc01 = wmma_bf16(A0l, B1h, acc01);

        acc10 = wmma_bf16(A1h, B0h, acc10);
        acc10 = wmma_bf16(A1h, B0l, acc10);
        acc10 = wmma_bf16(A1l, B0h, acc10);

        acc11 = wmma_bf16(A1h, B1h, acc11);
        acc11 = wmma_bf16(A1h, B1l, acc11);
        acc11 = wmma_bf16(A1l, B1h, acc11);
    }

    // C layout: VGPR r -> M = r + 8*hv, lane&15 -> N
    const float scale = 0.0625f;   // 1/sqrt(256)
    #pragma unroll
    for (int mt = 0; mt < 2; ++mt) {
        v8f accn0 = mt ? acc10 : acc00;
        v8f accn1 = mt ? acc11 : acc01;
        #pragma unroll
        for (int r = 0; r < 8; ++r) {
            int mrow = m0 + mt * 16 + r + hv * 8;
            float* o = out + ((size_t)b * HW_ + mrow) * HW_;
            o[n0 + l15]      = accn0[r] * scale;
            o[n0 + 16 + l15] = accn1[r] * scale;
        }
    }
}

// ---------- 2x2 average pool, stride 2 ----------
__global__ void pool2_kernel(const float* __restrict__ src, float* __restrict__ dst,
                             int Hi, int Wi, int totalOut) {
    int Ho = Hi >> 1, Wo = Wi >> 1;
    for (int idx = blockIdx.x * blockDim.x + threadIdx.x; idx < totalOut;
         idx += gridDim.x * blockDim.x) {
        int jo = idx % Wo;
        int t  = idx / Wo;
        int io = t % Ho;
        int n  = t / Ho;
        const float* s = src + (size_t)n * Hi * Wi + (size_t)(2 * io) * Wi + 2 * jo;
        dst[idx] = 0.25f * (s[0] + s[1] + s[Wi] + s[Wi + 1]);
    }
}

// ---------- windowed bilinear lookup (zero padding), one block per pixel ----------
__global__ void lookup_kernel(const float* __restrict__ coords,
                              const float* __restrict__ lvl0,
                              const float* __restrict__ lvl1,
                              const float* __restrict__ lvl2,
                              const float* __restrict__ lvl3,
                              float* __restrict__ out) {
    int n   = blockIdx.x;            // 0..NPIX-1
    int b   = n / HW_;
    int rem = n - b * HW_;
    int i   = rem / W_;
    int j   = rem - i * W_;
    float cx = coords[(size_t)(b * 2)     * HW_ + rem];
    float cy = coords[(size_t)(b * 2 + 1) * HW_ + rem];

    const float* bases[4] = { lvl0 + (size_t)n * HW_,
                              lvl1 + (size_t)n * (HW_ / 4),
                              lvl2 + (size_t)n * (HW_ / 16),
                              lvl3 + (size_t)n * (HW_ / 64) };

    for (int k = threadIdx.x; k < 4 * 81; k += blockDim.x) {
        int l  = k / 81;
        int kk = k - l * 81;
        float inv = 1.0f / (float)(1 << l);
        // RAFT quirk: dy-grid added to x, dx-grid added to y
        float x = cx * inv + (float)(kk / 9) - 4.0f;
        float y = cy * inv + (float)(kk % 9) - 4.0f;
        int Hl = H_ >> l, Wl = W_ >> l;
        const float* img = bases[l];

        float x0f = floorf(x), y0f = floorf(y);
        float wx = x - x0f, wy = y - y0f;
        int x0 = (int)x0f, y0 = (int)y0f;
        int x1 = x0 + 1,   y1 = y0 + 1;
        bool bx0 = (x0 >= 0) & (x0 < Wl), bx1 = (x1 >= 0) & (x1 < Wl);
        bool by0 = (y0 >= 0) & (y0 < Hl), by1 = (y1 >= 0) & (y1 < Hl);
        float v00 = (bx0 & by0) ? img[y0 * Wl + x0] : 0.0f;
        float v10 = (bx1 & by0) ? img[y0 * Wl + x1] : 0.0f;
        float v01 = (bx0 & by1) ? img[y1 * Wl + x0] : 0.0f;
        float v11 = (bx1 & by1) ? img[y1 * Wl + x1] : 0.0f;
        float v = v00 * (1.0f - wx) * (1.0f - wy) + v10 * wx * (1.0f - wy)
                + v01 * (1.0f - wx) * wy          + v11 * wx * wy;

        out[(((size_t)b * 324 + k) * H_ + i) * W_ + j] = v;
    }
}

extern "C" void kernel_launch(void* const* d_in, const int* in_sizes, int n_in,
                              void* d_out, int out_size, void* d_ws, size_t ws_size,
                              hipStream_t stream) {
    (void)in_sizes; (void)n_in; (void)out_size; (void)ws_size;
    const float* fmap1  = (const float*)d_in[0];
    const float* fmap2  = (const float*)d_in[1];
    const float* coords = (const float*)d_in[2];
    float* out = (float*)d_out;

    // workspace layout: 4 transposed bf16 maps (31.5 MB) then the pyramid (626.7 MB)
    const size_t S = (size_t)B_ * D_ * HW_;         // 3,932,160 elems per map
    unsigned short* f1hi = (unsigned short*)d_ws;
    unsigned short* f1lo = f1hi + S;
    unsigned short* f2hi = f1lo + S;
    unsigned short* f2lo = f2hi + S;
    float* lvl0 = (float*)(f2lo + S);
    float* lvl1 = lvl0 + (size_t)NPIX * HW_;
    float* lvl2 = lvl1 + (size_t)NPIX * (HW_ / 4);
    float* lvl3 = lvl2 + (size_t)NPIX * (HW_ / 16);

    const int totalMap = (int)S;
    split_t_kernel<<<(totalMap + 255) / 256, 256, 0, stream>>>(fmap1, f1hi, f1lo, totalMap);
    split_t_kernel<<<(totalMap + 255) / 256, 256, 0, stream>>>(fmap2, f2hi, f2lo, totalMap);

    dim3 ggrid(HW_ / 128, HW_ / 64, B_);   // (60, 120, 2)
    corr_wmma_kernel<<<ggrid, 256, 0, stream>>>(f1hi, f1lo, f2hi, f2lo, lvl0);

    int t1 = NPIX * (HW_ / 4);
    pool2_kernel<<<(t1 + 255) / 256, 256, 0, stream>>>(lvl0, lvl1, H_,     W_,     t1);
    int t2 = NPIX * (HW_ / 16);
    pool2_kernel<<<(t2 + 255) / 256, 256, 0, stream>>>(lvl1, lvl2, H_ / 2, W_ / 2, t2);
    int t3 = NPIX * (HW_ / 64);
    pool2_kernel<<<(t3 + 255) / 256, 256, 0, stream>>>(lvl2, lvl3, H_ / 4, W_ / 4, t3);

    lookup_kernel<<<NPIX, 128, 0, stream>>>(coords, lvl0, lvl1, lvl2, lvl3, out);
}
